// EMG_15101105013512
// MI455X (gfx1250) — matrix operand order repaired and checked by reading, not compiled, not run on hardware
//
#include <hip/hip_runtime.h>
#include <hip/hip_bf16.h>
#include <stdint.h>

// EMG 2-layer recurrent cell, B=32 T=1024 I=H=512, MI455X (gfx1250, wave32).
// Strategy: bf16 WMMA (v_wmma_f32_16x16x32_bf16) for all GEMMs, f32 state.
// Persistent cooperative kernel: 80 blocks x 128 thr (4 wave32) = 320 waves,
// one 16x16 WMMA tile per wave per phase, 3 grid barriers per timestep.
// Weights (3 MB bf16) stay resident in the 192 MB L2 for all 1024 steps.

#define EB 32
#define ET 1024
#define EI 512
#define EH 512
#define EG (2 * EH)
#define NBLOCKS 80
#define OUT_ELEMS ((size_t)EB * ET * 2 * EH)

typedef __attribute__((ext_vector_type(16))) __bf16 v16bf;
typedef __attribute__((ext_vector_type(8)))  __bf16 v8bf;
typedef __attribute__((ext_vector_type(8)))  float  v8f;
typedef __attribute__((ext_vector_type(4)))  float  v4f;

// ---------------- workspace layout (bytes) ----------------
#define WS_WT_IN0 0u          // 512*512  bf16 = 524288
#define WS_WT_H0  524288u     // 1024*512 bf16 = 1048576
#define WS_WT_IN1 1572864u    // 512*512  bf16
#define WS_WT_H1  2097152u    // 1024*512 bf16
#define WS_C0     3145728u    // 32*512 f32
#define WS_C1     3211264u
#define WS_M0BF   3276800u    // 32*512 bf16
#define WS_M1BF   3309568u
#define WS_G0     3342336u    // 32*1024 f32
#define WS_G1     3473408u
#define WS_CNT    3604480u    // barrier counter
#define WS_ZERO_WORDS 49152   // (c0,c1,m0bf,m1bf) region in 32-bit words

__device__ __forceinline__ float sigmoidf_(float x) {
  return 1.0f / (1.0f + __expf(-x));
}

// ---- WMMA operand loaders (wave32 lane layouts per CDNA5 ISA 7.12.2) ----
// A (16x32 bf16): lane L holds row M=L&15; K-chunks {0..7} and {16..23}
// shifted by 8 for the upper half-wave.
__device__ __forceinline__ v16bf load_a_bf(const __bf16* base, int ldk,
                                           int mbase, int kc) {
  int lane = threadIdx.x & 31;
  int row  = mbase + (lane & 15);
  int ko   = (lane & 16) ? 8 : 0;
  const __bf16* p = base + (size_t)row * ldk + kc + ko;
  v8bf lo = *(const v8bf*)p;
  v8bf hi = *(const v8bf*)(p + 16);
  v16bf v;
#pragma unroll
  for (int i = 0; i < 8; ++i) { v[i] = lo[i]; v[i + 8] = hi[i]; }
  return v;
}

// Same as load_a_bf but source is f32 (the x tensor), converted in-register.
__device__ __forceinline__ v16bf load_a_f32(const float* base, int ldk,
                                            int mbase, int kc) {
  int lane = threadIdx.x & 31;
  int row  = mbase + (lane & 15);
  int ko   = (lane & 16) ? 8 : 0;
  const float* p = base + (size_t)row * ldk + kc + ko;
  v4f a0 = *(const v4f*)(p + 0);
  v4f a1 = *(const v4f*)(p + 4);
  v4f b0 = *(const v4f*)(p + 16);
  v4f b1 = *(const v4f*)(p + 20);
  v16bf v;
#pragma unroll
  for (int i = 0; i < 4; ++i) {
    v[i]      = (__bf16)a0[i];
    v[i + 4]  = (__bf16)a1[i];
    v[i + 8]  = (__bf16)b0[i];
    v[i + 12] = (__bf16)b1[i];
  }
  return v;
}

// B (32x16 bf16), weights pre-transposed to [N][K]: lane L holds column
// N=L&15; K-chunks {0..7} and {16..23} shifted by 8 for the upper half-wave.
__device__ __forceinline__ v16bf load_b_bf(const __bf16* Wt, int ldk,
                                           int nbase, int kc) {
  int lane = threadIdx.x & 31;
  int n    = nbase + (lane & 15);
  int ko   = (lane & 16) ? 8 : 0;
  const __bf16* p = Wt + (size_t)n * ldk + kc + ko;
  v8bf lo = *(const v8bf*)p;
  v8bf hi = *(const v8bf*)(p + 16);
  v16bf v;
#pragma unroll
  for (int i = 0; i < 8; ++i) { v[i] = lo[i]; v[i + 8] = hi[i]; }
  return v;
}

// Monotonic wrap-safe grid barrier (all 80 blocks resident).
__device__ __forceinline__ void grid_barrier(unsigned int* cnt) {
  __syncthreads();
  if (threadIdx.x == 0) {
    __threadfence();                      // release prior global writes
    unsigned int arrived = atomicAdd(cnt, 1u) + 1u;
    unsigned int target  = ((arrived + NBLOCKS - 1u) / NBLOCKS) * NBLOCKS;
    while ((int)(*(volatile unsigned int*)cnt - target) < 0)
      __builtin_amdgcn_s_sleep(1);
    __threadfence();                      // acquire others' writes
  }
  __syncthreads();
}

// ---------------- kernels ----------------
__global__ void emg_init(unsigned int* zero_words, unsigned int* cnt) {
  int id = blockIdx.x * blockDim.x + threadIdx.x;
  if (id < WS_ZERO_WORDS) zero_words[id] = 0u;
  if (id == 0) *cnt = 0u;
}

// W[K][N] f32  ->  Wt[N][K] bf16
__global__ void emg_transpose_bf16(const float* __restrict__ W,
                                   __bf16* __restrict__ Wt, int K, int N) {
  int id = blockIdx.x * blockDim.x + threadIdx.x;
  if (id >= K * N) return;
  int n = id / K;
  int k = id - n * K;
  Wt[id] = (__bf16)W[(size_t)k * N + n];
}

__global__ __launch_bounds__(128) void emg_scan(
    const float* __restrict__ x,
    const float* __restrict__ b_in0, const float* __restrict__ b_h0,
    const float* __restrict__ b_in1, const float* __restrict__ b_h1,
    const __bf16* __restrict__ Wt_in0, const __bf16* __restrict__ Wt_h0,
    const __bf16* __restrict__ Wt_in1, const __bf16* __restrict__ Wt_h1,
    float* __restrict__ c0, float* __restrict__ c1,
    __bf16* __restrict__ m0bf, __bf16* __restrict__ m1bf,
    float* __restrict__ g0, float* __restrict__ g1,
    float* __restrict__ out, unsigned int* cnt) {
  const int wave = threadIdx.x >> 5;
  const int gw   = blockIdx.x * 4 + wave;   // 0..319
  const int lane = threadIdx.x & 31;
  const int rsh  = (lane & 16) ? 8 : 0;     // C/D row shift for upper half-wave
  const int ncol = lane & 15;               // C/D column within tile

  for (int t = 0; t < ET; ++t) {
    v8f ti_acc = {0.f, 0.f, 0.f, 0.f, 0.f, 0.f, 0.f, 0.f};

    // ---- Phase A: gates0 (waves 0..127), gates1 (128..255), ti0 (256..319)
    if (gw < 128) {
      int mi = gw >> 6, nj = gw & 63;
      v8f acc = {0.f, 0.f, 0.f, 0.f, 0.f, 0.f, 0.f, 0.f};
#pragma unroll 4
      for (int kc = 0; kc < EH; kc += 32) {
        __builtin_prefetch(Wt_h0 + (size_t)(nj * 16 + ncol) * EH + kc + 32, 0, 1);
        v16bf a = load_a_bf(m0bf, EH, mi * 16, kc);
        v16bf b = load_b_bf(Wt_h0, EH, nj * 16, kc);
        acc = __builtin_amdgcn_wmma_f32_16x16x32_bf16(
            false, a, false, b, (short)0, acc, false, false);
      }
      int rbase = mi * 16 + rsh, n = nj * 16 + ncol;
#pragma unroll
      for (int e = 0; e < 8; ++e) g0[(rbase + e) * EG + n] = acc[e];
    } else if (gw < 256) {
      int gg = gw - 128, mi = gg >> 6, nj = gg & 63;
      v8f acc = {0.f, 0.f, 0.f, 0.f, 0.f, 0.f, 0.f, 0.f};
#pragma unroll 4
      for (int kc = 0; kc < EH; kc += 32) {
        __builtin_prefetch(Wt_h1 + (size_t)(nj * 16 + ncol) * EH + kc + 32, 0, 1);
        v16bf a = load_a_bf(m1bf, EH, mi * 16, kc);
        v16bf b = load_b_bf(Wt_h1, EH, nj * 16, kc);
        acc = __builtin_amdgcn_wmma_f32_16x16x32_bf16(
            false, a, false, b, (short)0, acc, false, false);
      }
      int rbase = mi * 16 + rsh, n = nj * 16 + ncol;
#pragma unroll
      for (int e = 0; e < 8; ++e) g1[(rbase + e) * EG + n] = acc[e];
    } else {
      int gg = gw - 256, mi = gg >> 5, nj = gg & 31;
      const float* xt = x + (size_t)t * EI;   // row stride T*I over batch
#pragma unroll 4
      for (int kc = 0; kc < EI; kc += 32) {
        v16bf a = load_a_f32(xt, ET * EI, mi * 16, kc);
        v16bf b = load_b_bf(Wt_in0, EI, nj * 16, kc);
        ti_acc = __builtin_amdgcn_wmma_f32_16x16x32_bf16(
            false, a, false, b, (short)0, ti_acc, false, false);
      }
    }
    grid_barrier(cnt);

    // ---- Phase B: layer-0 cell update (ti0-owning waves) ----
    if (gw >= 256) {
      int gg = gw - 256, mi = gg >> 5, nj = gg & 31;
      int rbase = mi * 16 + rsh, n = nj * 16 + ncol;
      float bi  = b_in0[n];
      float bhr = b_h0[n], bhm = b_h0[EH + n];
#pragma unroll
      for (int e = 0; e < 8; ++e) {
        int br = rbase + e;
        float ti     = ti_acc[e] + bi;
        float retain = sigmoidf_(g0[br * EG + n] + bhr);
        float merge  = sigmoidf_(g0[br * EG + EH + n] + bhm);
        float cp = c0[br * EH + n];
        float cn = tanhf(cp + retain * ti);
        float mn = (1.0f - merge) * ti + merge * cn;
        c0[br * EH + n]   = cn;
        m0bf[br * EH + n] = (__bf16)mn;
        size_t o = (((size_t)br * ET + t) * 2 + 0) * EH + n;
        out[o] = mn;
        out[OUT_ELEMS + o] = mn;
      }
    }
    grid_barrier(cnt);

    // ---- Phase C: ti1 GEMM + fused layer-1 cell update (waves 0..63) ----
    if (gw < 64) {
      int mi = gw >> 5, nj = gw & 31;
      v8f acc = {0.f, 0.f, 0.f, 0.f, 0.f, 0.f, 0.f, 0.f};
#pragma unroll 4
      for (int kc = 0; kc < EH; kc += 32) {
        v16bf a = load_a_bf(m0bf, EH, mi * 16, kc);
        v16bf b = load_b_bf(Wt_in1, EH, nj * 16, kc);
        acc = __builtin_amdgcn_wmma_f32_16x16x32_bf16(
            false, a, false, b, (short)0, acc, false, false);
      }
      int rbase = mi * 16 + rsh, n = nj * 16 + ncol;
      float bi  = b_in1[n];
      float bhr = b_h1[n], bhm = b_h1[EH + n];
#pragma unroll
      for (int e = 0; e < 8; ++e) {
        int br = rbase + e;
        float ti     = acc[e] + bi;
        float retain = sigmoidf_(g1[br * EG + n] + bhr);
        float merge  = sigmoidf_(g1[br * EG + EH + n] + bhm);
        float cp = c1[br * EH + n];
        float cn = tanhf(cp + retain * ti);
        float mn = (1.0f - merge) * ti + merge * cn;
        c1[br * EH + n]   = cn;
        m1bf[br * EH + n] = (__bf16)mn;
        size_t o = (((size_t)br * ET + t) * 2 + 1) * EH + n;
        out[o] = mn;
        out[OUT_ELEMS + o] = mn;
      }
    }
    grid_barrier(cnt);
  }
}

// con = broadcast(final c over T):  out2[b][t][l][h] = c_l[b][h]
__global__ void emg_con(const float* __restrict__ c0,
                        const float* __restrict__ c1,
                        float* __restrict__ out2) {
  size_t id = (size_t)blockIdx.x * blockDim.x + threadIdx.x;
  int h = (int)(id & (EH - 1));
  int l = (int)((id >> 9) & 1);
  int b = (int)(id >> 20);
  out2[id] = l ? c1[b * EH + h] : c0[b * EH + h];
}

extern "C" void kernel_launch(void* const* d_in, const int* in_sizes, int n_in,
                              void* d_out, int out_size, void* d_ws,
                              size_t ws_size, hipStream_t stream) {
  const float* x     = (const float*)d_in[0];
  const float* W_in0 = (const float*)d_in[1];
  const float* b_in0 = (const float*)d_in[2];
  const float* W_h0  = (const float*)d_in[3];
  const float* b_h0  = (const float*)d_in[4];
  const float* W_in1 = (const float*)d_in[5];
  const float* b_in1 = (const float*)d_in[6];
  const float* W_h1  = (const float*)d_in[7];
  const float* b_h1  = (const float*)d_in[8];
  float* out = (float*)d_out;

  char* ws = (char*)d_ws;
  __bf16* Wt_in0 = (__bf16*)(ws + WS_WT_IN0);
  __bf16* Wt_h0  = (__bf16*)(ws + WS_WT_H0);
  __bf16* Wt_in1 = (__bf16*)(ws + WS_WT_IN1);
  __bf16* Wt_h1  = (__bf16*)(ws + WS_WT_H1);
  float*  c0     = (float*)(ws + WS_C0);
  float*  c1     = (float*)(ws + WS_C1);
  __bf16* m0bf   = (__bf16*)(ws + WS_M0BF);
  __bf16* m1bf   = (__bf16*)(ws + WS_M1BF);
  float*  g0     = (float*)(ws + WS_G0);
  float*  g1     = (float*)(ws + WS_G1);
  unsigned int* cnt = (unsigned int*)(ws + WS_CNT);

  // Re-init state + barrier every call (graph replay determinism).
  emg_init<<<(WS_ZERO_WORDS + 255) / 256, 256, 0, stream>>>(
      (unsigned int*)(ws + WS_C0), cnt);

  // One-time (per call) weight transpose + bf16 convert: Wt[N][K].
  emg_transpose_bf16<<<(512 * 512) / 256, 256, 0, stream>>>(W_in0, Wt_in0, 512, 512);
  emg_transpose_bf16<<<(512 * 1024) / 256, 256, 0, stream>>>(W_h0, Wt_h0, 512, 1024);
  emg_transpose_bf16<<<(512 * 512) / 256, 256, 0, stream>>>(W_in1, Wt_in1, 512, 512);
  emg_transpose_bf16<<<(512 * 1024) / 256, 256, 0, stream>>>(W_h1, Wt_h1, 512, 1024);

  // Persistent cooperative scan: 80 blocks x 128 threads (320 wave32s).
  emg_scan<<<NBLOCKS, 128, 0, stream>>>(x, b_in0, b_h0, b_in1, b_h1,
                                        Wt_in0, Wt_h0, Wt_in1, Wt_h1,
                                        c0, c1, m0bf, m1bf, g0, g1, out, cnt);

  // Broadcast final cell states into the third output tensor.
  emg_con<<<(unsigned)(OUT_ELEMS / 256), 256, 0, stream>>>(
      c0, c1, out + 2 * OUT_ELEMS);
}
